// TdLstm_41910290874950
// MI455X (gfx1250) — matrix operand order repaired and checked by reading, not compiled, not run on hardware
//
#include <hip/hip_runtime.h>

// ---------------------------------------------------------------------------
// Siamese LSTM (TF BasicLSTMCell) for MI455X / gfx1250, wave32 + WMMA f16.
//
// Pipeline:
//   1) pack_w / pack_bias : fp32 W -> f16 WMMA B-fragments (K padded 320,
//      per-gate cols padded 304 -> 1216), zero-padded.
//   2) per 64-step chunk:
//      a) xproj_kernel : gather embeddings + GEMM  x_t @ Wx + b  (parallel,
//         v_wmma_f32_16x16x32_f16), result stored in C/D-fragment order.
//      b) lstm_kernel  : 8 WGs per side; each WG stages its Wh column slice
//         (<=120KB) into LDS via GLOBAL_LOAD_ASYNC_TO_LDS_B128 once, then
//         runs 64 recurrent steps.  h ping-pongs through global f16 buffers;
//         c and own-column h stay in VGPRs; per-step global ticket barrier.
//   3) dense_softmax : 128 threads, logits/probs/loss.
// ---------------------------------------------------------------------------

typedef _Float16 v16h __attribute__((ext_vector_type(16)));
typedef _Float16 v8h  __attribute__((ext_vector_type(8)));
typedef float    v8f  __attribute__((ext_vector_type(8)));

#define DEVI static __device__ __forceinline__

// problem sizes
constexpr int BB   = 128;     // batch
constexpr int TT   = 256;     // seq len
constexpr int EMB  = 300;
constexpr int HID  = 300;
constexpr int KT   = 10;      // K tiles (K padded to 320)
constexpr int KPAD = 320;
constexpr int NTIL = 19;      // 16-col tiles per gate (304/16)
constexpr int HPAD = 304;
constexpr int CTIL = 76;      // 4*19 total col tiles
constexpr int GPAD = 1216;    // 4*304 padded gate cols
constexpr int CH   = 64;      // timesteps per chunk
constexpr int NCHUNK = 4;
constexpr int FRH  = 512;     // halves per 32x16 B-fragment (1KB)
constexpr int XFR  = 256;     // halves per 16x16 C/D fragment (512B)
constexpr int NWG  = 8;       // workgroups per side in recurrent kernel

constexpr size_t WFH = (size_t)KT * CTIL * FRH;          // halves per packed W

// workspace layout (bytes)
constexpr size_t O_WXF  = 0;
constexpr size_t O_WHF  = O_WXF  + 2 * WFH * 2;
constexpr size_t O_BIAS = O_WHF  + 2 * WFH * 2;
constexpr size_t SZ_BIAS= 2 * GPAD * 4;
constexpr size_t O_HBUF = O_BIAS + SZ_BIAS;              // [ping][side][128][320] f16
constexpr size_t SZ_HBUF= (size_t)2 * 2 * BB * KPAD * 2;
constexpr size_t O_CST  = O_HBUF + SZ_HBUF;              // [side][128][304] f32
constexpr size_t SZ_CST = (size_t)2 * BB * HPAD * 4;
constexpr size_t O_BAR  = O_CST + SZ_CST;                // [side][64] u32
constexpr size_t SZ_BAR = 2 * CH * 4;
// xproj, C/D fragment order: [side][row_tile(CH*8)][ct(76)][lane(32)][8] f16
constexpr size_t O_XP   = O_BAR + SZ_BAR;
constexpr size_t SZ_XP  = (size_t)2 * CH * 8 * CTIL * XFR * 2;

DEVI v8f wmma16(v16h a, v16h b, v8f c) {
  return __builtin_amdgcn_wmma_f32_16x16x32_f16(
      /*neg_a=*/false, a, /*neg_b=*/false, b,
      /*c_mod=*/(short)0, c, /*reuse_a=*/false, /*reuse_b=*/false);
}

DEVI v16h cat16(v8h lo, v8h hi) {
  return __builtin_shufflevector(lo, hi, 0, 1, 2, 3, 4, 5, 6, 7,
                                 8, 9, 10, 11, 12, 13, 14, 15);
}

DEVI float sigmf(float x) { return 1.0f / (1.0f + __expf(-x)); }

// async global->LDS copy of 16 bytes per lane (CDNA5, tracked by ASYNCcnt)
DEVI void async_g2l_b128(unsigned lds_byte_addr, const void* gptr) {
  asm volatile("global_load_async_to_lds_b128 %0, %1, off"
               :
               : "v"(lds_byte_addr), "v"((unsigned long long)gptr)
               : "memory");
}
DEVI void wait_async0() {
#if __has_builtin(__builtin_amdgcn_s_wait_asynccnt)
  __builtin_amdgcn_s_wait_asynccnt(0);
#else
  asm volatile("s_wait_asynccnt 0x0" ::: "memory");
#endif
}

// -------------------------------------------------------------------------
// Pack W (fp32 [600][1200]) into f16 B-fragments, zero padded.
// Fragment layout: [kt][ct][lane][16 halves]; lane=N%16 (+16 = K upper half);
// half j within lane -> K_local = (lane/16)*8 + j (j<8), 16 + (lane/16)*8 + (j-8) (j>=8).
// -------------------------------------------------------------------------
__global__ __launch_bounds__(256) void pack_w(const float* __restrict__ Wl,
                                              const float* __restrict__ Wr,
                                              _Float16* __restrict__ wxf,
                                              _Float16* __restrict__ whf) {
  const int PER = (int)WFH;  // 389120
  int idx = blockIdx.x * blockDim.x + threadIdx.x;
  if (idx >= 4 * PER) return;
  int sm   = idx / PER;           // side*2 + mat
  int side = sm >> 1, mat = sm & 1;
  int r    = idx % PER;
  int kt   = r / (CTIL * FRH);
  int r2   = r % (CTIL * FRH);
  int ct   = r2 / FRH;
  int q    = r2 % FRH;
  int lane = q >> 4, j = q & 15;
  int kloc = ((lane >> 4) << 3) + (j & 7) + ((j >> 3) << 4);  // table order
  int g    = ct / NTIL;
  int n    = (ct % NTIL) * 16 + (lane & 15);
  int k    = kt * 32 + kloc;
  float val = 0.0f;
  if (n < HID && k < (mat ? HID : EMB)) {
    const float* W = side ? Wr : Wl;
    int row = mat ? (EMB + k) : k;
    val = W[row * 1200 + g * 300 + n];
  }
  _Float16* dst = mat ? whf : wxf;
  dst[(size_t)side * PER + ((size_t)kt * CTIL + ct) * FRH + q] = (_Float16)val;
}

__global__ __launch_bounds__(256) void pack_bias(const float* __restrict__ bl,
                                                 const float* __restrict__ br,
                                                 float* __restrict__ biasp) {
  int idx = blockIdx.x * blockDim.x + threadIdx.x;
  if (idx >= 2 * GPAD) return;
  int side = idx / GPAD, col = idx % GPAD;
  int g = col / HPAD, n = col % HPAD;
  const float* b = side ? br : bl;
  biasp[idx] = (n < HID) ? b[g * 300 + n] : 0.0f;
}

// -------------------------------------------------------------------------
// xproj = gather(embedding) @ Wx + b    for one 64-step chunk, both sides.
// grid (CH*128/16, 2), block 256 (8 waves). Each WG: one 16-row strip.
// Output in C/D fragment order so the recurrent kernel reads v8h directly.
// -------------------------------------------------------------------------
__global__ __launch_bounds__(256) void xproj_kernel(
    const int* __restrict__ lids, const int* __restrict__ rids,
    const float* __restrict__ emb, const _Float16* __restrict__ wxf,
    const float* __restrict__ biasp, _Float16* __restrict__ xp, int chunk) {
  __shared__ _Float16 Ab[16 * KPAD];  // 10KB: 16 rows x K(320) f16
  const int side = blockIdx.y;
  const int mt   = blockIdx.x;
  const int tid  = threadIdx.x;
  const int* ids = side ? rids : lids;

  for (int i = tid; i < 16 * KPAD; i += 256) {
    int rr = i / KPAD, k = i % KPAD;
    int rg = mt * 16 + rr;            // chunk row = tl*128 + b
    int tl = rg >> 7, b = rg & 127;
    float v = 0.0f;
    if (k < EMB) {
      int id = ids[b * TT + chunk * CH + tl];
      v = emb[(size_t)id * EMB + k];
    }
    Ab[i] = (_Float16)v;
  }
  __syncthreads();

  const int wave = tid >> 5, lane = tid & 31;
  const int lm = lane & 15, lh = lane >> 4;
  const _Float16* wx = wxf + (size_t)side * WFH;
  _Float16* xps = xp + (size_t)side * CH * 8 * CTIL * XFR;

  for (int ct = wave; ct < CTIL; ct += 8) {   // wave-uniform loop, EXEC full
    int colp = ct * 16 + lm;
    float bv = biasp[side * GPAD + colp];
    v8f acc = {bv, bv, bv, bv, bv, bv, bv, bv};
#pragma unroll
    for (int kt = 0; kt < KT; ++kt) {
      const _Float16* ap = &Ab[lm * KPAD + kt * 32 + (lh << 3)];
      v16h a = cat16(*(const v8h*)ap, *(const v8h*)(ap + 16));
      v16h b = *(const v16h*)(wx + ((size_t)kt * CTIL + ct) * FRH + lane * 16);
      acc = wmma16(a, b, acc);
    }
    v8h hv;
#pragma unroll
    for (int v = 0; v < 8; ++v) hv[v] = (_Float16)acc[v];
    *(v8h*)(xps + ((size_t)mt * CTIL + ct) * XFR + lane * 8) = hv;
  }
}

// -------------------------------------------------------------------------
// Recurrent chunk.  grid (NWG, 2 sides), block 512 (16 waves).
// WG 'wg' owns ntiles {wg, wg+8, wg+16} (<19); stages Wh slice in LDS once
// via async global->LDS b128.  Per step: z = xproj + h@Wh ; gates; mask;
// h ping-pong via global; c and own-column h live in VGPRs.
// -------------------------------------------------------------------------
__global__ __launch_bounds__(512) void lstm_kernel(
    const int* __restrict__ llen, const int* __restrict__ rlen,
    const _Float16* __restrict__ whf, const _Float16* __restrict__ xp,
    _Float16* __restrict__ hbuf, float* __restrict__ cst,
    unsigned int* __restrict__ bar, int chunk) {
  extern __shared__ _Float16 smem[];  // up to 3*4*10 frags = 120KB
  const int side = blockIdx.y, wg = blockIdx.x;
  const int tid = threadIdx.x;
  const int nOwn = (wg < 3) ? 3 : 2;  // wg + 16 < 19 ?

  {  // async-stage Wh slice: smem[((own*4+g)*10+kt)*512 ..], 16B per lane/iter
    const char* srcb = (const char*)(whf + (size_t)side * WFH);
    unsigned ldsbase = (unsigned)(unsigned long long)(void*)smem;
    int chunks = nOwn * 40 * 64;  // 16B chunks; multiple of 512 -> EXEC full
    for (int i = tid; i < chunks; i += 512) {
      int own = i / (40 * 64);
      int rem = i % (40 * 64);
      int g   = rem / (10 * 64);
      int r2  = rem % (10 * 64);
      int kt  = r2 >> 6;
      int q   = r2 & 63;
      int nt  = wg + 8 * own;
      const char* src =
          srcb + (((size_t)kt * CTIL + (g * NTIL + nt)) * (FRH * 2) + q * 16);
      async_g2l_b128(ldsbase + (unsigned)i * 16, src);
    }
    wait_async0();
  }
  __syncthreads();

  const int wave = tid >> 5, lane = tid & 31;
  const int lm = lane & 15, lh = lane >> 4;
  const int* lens = side ? rlen : llen;
  const int ntot = nOwn * 8;
  const int tA = wave, tB = wave + 16;      // tile ids (own = t/8, mtile = t%8)
  const bool hasA = (tA < ntot), hasB = (tB < ntot);   // wave-uniform

  float* cs = cst + (size_t)side * BB * HPAD;
  const _Float16* xps = xp + (size_t)side * CH * 8 * CTIL * XFR;
  _Float16* h0 = hbuf + (size_t)(0 * 2 + side) * BB * KPAD;
  _Float16* h1 = hbuf + (size_t)(1 * 2 + side) * BB * KPAD;
  unsigned int* bars = bar + side * CH;

  auto loadC = [&](int tile) {
    int own = tile >> 3, mtile = tile & 7;
    int col = (wg + 8 * own) * 16 + lm;
    v8f c = {0, 0, 0, 0, 0, 0, 0, 0};
#pragma unroll
    for (int v = 0; v < 8; ++v)
      c[v] = cs[(size_t)(mtile * 16 + v + (lh << 3)) * HPAD + col];
    return c;
  };
  // own-column h state (chunk start parity is always ping 0: chunk*64 even)
  auto loadH = [&](int tile) {
    int own = tile >> 3, mtile = tile & 7;
    int col = (wg + 8 * own) * 16 + lm;
    v8f h = {0, 0, 0, 0, 0, 0, 0, 0};
#pragma unroll
    for (int v = 0; v < 8; ++v)
      h[v] = (float)h0[(size_t)(mtile * 16 + v + (lh << 3)) * KPAD + col];
    return h;
  };
  v8f cA = hasA ? loadC(tA) : v8f{0, 0, 0, 0, 0, 0, 0, 0};
  v8f cB = hasB ? loadC(tB) : v8f{0, 0, 0, 0, 0, 0, 0, 0};
  v8f hA = hasA ? loadH(tA) : v8f{0, 0, 0, 0, 0, 0, 0, 0};
  v8f hB = hasB ? loadH(tB) : v8f{0, 0, 0, 0, 0, 0, 0, 0};

  for (int tl = 0; tl < CH; ++tl) {
    const int tg = chunk * CH + tl;
    const _Float16* hr = (tg & 1) ? h1 : h0;
    _Float16* hw       = (tg & 1) ? h0 : h1;

    auto doTile = [&](int tile, v8f& creg, v8f& hreg) {
      int own = tile >> 3, mtile = tile & 7;
      int nt = wg + 8 * own;
      int colh = nt * 16 + lm;                         // hidden col (0..303)
      // accumulator init: xproj fragments (C/D order), one v8h per gate
      size_t mtg = (size_t)(tl * 8 + mtile) * CTIL;
      v8h xi = *(const v8h*)(xps + (mtg + 0 * NTIL + nt) * XFR + lane * 8);
      v8h xj = *(const v8h*)(xps + (mtg + 1 * NTIL + nt) * XFR + lane * 8);
      v8h xf = *(const v8h*)(xps + (mtg + 2 * NTIL + nt) * XFR + lane * 8);
      v8h xo = *(const v8h*)(xps + (mtg + 3 * NTIL + nt) * XFR + lane * 8);
      v8f ai, aj, af, ao;
#pragma unroll
      for (int v = 0; v < 8; ++v) {
        ai[v] = (float)xi[v];
        aj[v] = (float)xj[v];
        af[v] = (float)xf[v];
        ao[v] = (float)xo[v];
      }
#pragma unroll
      for (int kt = 0; kt < KT; ++kt) {
        const _Float16* ap =
            hr + (size_t)(mtile * 16 + lm) * KPAD + kt * 32 + (lh << 3);
        v16h a = cat16(*(const v8h*)ap, *(const v8h*)(ap + 16));
        const _Float16* bb = smem + ((size_t)(own * 4) * 10 + kt) * FRH;
        v16h b0 = *(const v16h*)(bb + 0 * 10 * FRH + lane * 16);
        v16h b1 = *(const v16h*)(bb + 1 * 10 * FRH + lane * 16);
        v16h b2 = *(const v16h*)(bb + 2 * 10 * FRH + lane * 16);
        v16h b3 = *(const v16h*)(bb + 3 * 10 * FRH + lane * 16);
        ai = wmma16(a, b0, ai);
        aj = wmma16(a, b1, aj);
        af = wmma16(a, b2, af);
        ao = wmma16(a, b3, ao);
      }
#pragma unroll
      for (int v = 0; v < 8; ++v) {
        int row = mtile * 16 + v + (lh << 3);
        float c_old = creg[v];
        float c_new = sigmf(af[v] + 1.0f) * c_old + sigmf(ai[v]) * tanhf(aj[v]);
        float h_new = sigmf(ao[v]) * tanhf(c_new);
        bool m = tg < lens[row];
        float co = m ? c_new : c_old;
        float ho = m ? h_new : hreg[v];
        creg[v] = co;
        hreg[v] = ho;
        hw[(size_t)row * KPAD + colh] = (_Float16)ho;
      }
    };
    if (hasA) doTile(tA, cA, hA);
    if (hasB) doTile(tB, cB, hB);

    if (tl + 1 < CH)  // warm L2/L0 for next step's activations
      __builtin_prefetch(
          &xps[(size_t)(tl + 1) * 8 * CTIL * XFR + (size_t)tid * 256], 0, 1);

    // per-step device barrier across the 8 WGs of this side (ticket counter)
    __threadfence();
    __syncthreads();
    if (tid == 0) {
      __hip_atomic_fetch_add(&bars[tl], 1u, __ATOMIC_ACQ_REL,
                             __HIP_MEMORY_SCOPE_AGENT);
      while (__hip_atomic_load(&bars[tl], __ATOMIC_ACQUIRE,
                               __HIP_MEMORY_SCOPE_AGENT) < (unsigned)NWG)
        __builtin_amdgcn_s_sleep(1);
    }
    __syncthreads();
    __threadfence();
  }

  auto storeC = [&](int tile, const v8f& creg) {
    int own = tile >> 3, mtile = tile & 7;
    int col = (wg + 8 * own) * 16 + lm;
#pragma unroll
    for (int v = 0; v < 8; ++v)
      cs[(size_t)(mtile * 16 + v + (lh << 3)) * HPAD + col] = creg[v];
  };
  if (hasA) storeC(tA, cA);
  if (hasB) storeC(tB, cB);
}

// -------------------------------------------------------------------------
// Dense head + softmax + NLL loss.  Final h is in ping buffer 0 (T even).
// -------------------------------------------------------------------------
__global__ __launch_bounds__(128) void dense_kernel(
    const _Float16* __restrict__ hbuf, const float* __restrict__ Wd,
    const float* __restrict__ bd, const int* __restrict__ labels,
    float* __restrict__ out) {
  __shared__ float red[128];
  int b = threadIdx.x;
  const _Float16* hl = hbuf + (size_t)(0 * 2 + 0) * BB * KPAD + (size_t)b * KPAD;
  const _Float16* hr = hbuf + (size_t)(0 * 2 + 1) * BB * KPAD + (size_t)b * KPAD;
  float l0 = bd[0], l1 = bd[1], l2 = bd[2];
  for (int e = 0; e < HID; ++e) {
    float f = (float)hl[e];
    l0 += f * Wd[e * 3 + 0];
    l1 += f * Wd[e * 3 + 1];
    l2 += f * Wd[e * 3 + 2];
  }
  for (int e = 0; e < HID; ++e) {
    float f = (float)hr[e];
    l0 += f * Wd[(300 + e) * 3 + 0];
    l1 += f * Wd[(300 + e) * 3 + 1];
    l2 += f * Wd[(300 + e) * 3 + 2];
  }
  float m = fmaxf(l0, fmaxf(l1, l2));
  float e0 = __expf(l0 - m), e1 = __expf(l1 - m), e2 = __expf(l2 - m);
  float s = e0 + e1 + e2, inv = 1.0f / s;
  out[b * 3 + 0] = l0;
  out[b * 3 + 1] = l1;
  out[b * 3 + 2] = l2;
  out[384 + b * 3 + 0] = e0 * inv;
  out[384 + b * 3 + 1] = e1 * inv;
  out[384 + b * 3 + 2] = e2 * inv;
  float lse = m + __logf(s);
  int lab = labels[b];
  float lgl = (lab == 0) ? l0 : (lab == 1) ? l1 : l2;
  red[b] = lse - lgl;  // -log p[label]
  __syncthreads();
  for (int st = 64; st > 0; st >>= 1) {
    if (b < st) red[b] += red[b + st];
    __syncthreads();
  }
  if (b == 0) out[768] = red[0] / 128.0f;
}

// -------------------------------------------------------------------------
extern "C" void kernel_launch(void* const* d_in, const int* in_sizes, int n_in,
                              void* d_out, int out_size, void* d_ws,
                              size_t ws_size, hipStream_t stream) {
  const int*   lids = (const int*)d_in[0];
  const int*   rids = (const int*)d_in[1];
  const int*   llen = (const int*)d_in[2];
  const int*   rlen = (const int*)d_in[3];
  const int*   labs = (const int*)d_in[4];
  const float* emb  = (const float*)d_in[5];
  const float* Wl   = (const float*)d_in[6];
  const float* bl   = (const float*)d_in[7];
  const float* Wr   = (const float*)d_in[8];
  const float* br   = (const float*)d_in[9];
  const float* Wd   = (const float*)d_in[10];
  const float* bd   = (const float*)d_in[11];
  float* out = (float*)d_out;

  char* ws = (char*)d_ws;
  _Float16* wxf   = (_Float16*)(ws + O_WXF);
  _Float16* whf   = (_Float16*)(ws + O_WHF);
  float*    biasp = (float*)(ws + O_BIAS);
  _Float16* hbuf  = (_Float16*)(ws + O_HBUF);
  float*    cstp  = (float*)(ws + O_CST);
  unsigned int* barp = (unsigned int*)(ws + O_BAR);
  _Float16* xph   = (_Float16*)(ws + O_XP);

  const int lstm_lds = 3 * 4 * 10 * FRH * (int)sizeof(_Float16);  // 120KB
  (void)hipFuncSetAttribute((const void*)lstm_kernel,
                            hipFuncAttributeMaxDynamicSharedMemorySize,
                            lstm_lds);

  {  // pack weights + bias into padded f16 fragments
    int total = 4 * (int)WFH;
    pack_w<<<(total + 255) / 256, 256, 0, stream>>>(Wl, Wr, wxf, whf);
    pack_bias<<<(2 * GPAD + 255) / 256, 256, 0, stream>>>(bl, br, biasp);
  }
  // zero h ping-pong buffers + c state (contiguous region)
  hipMemsetAsync(ws + O_HBUF, 0, SZ_HBUF + SZ_CST, stream);

  for (int chunk = 0; chunk < NCHUNK; ++chunk) {
    hipMemsetAsync(ws + O_BAR, 0, SZ_BAR, stream);
    xproj_kernel<<<dim3(CH * BB / 16, 2), 256, 0, stream>>>(
        lids, rids, emb, wxf, biasp, xph, chunk);
    lstm_kernel<<<dim3(NWG, 2), 512, lstm_lds, stream>>>(
        llen, rlen, whf, xph, hbuf, cstp, barp, chunk);
  }
  dense_kernel<<<1, 128, 0, stream>>>(hbuf, Wd, bd, labs, out);
}
